// MoELayer_23227183137053
// MI455X (gfx1250) — matrix operand order repaired and checked by reading, not compiled, not run on hardware
//
#include <hip/hip_runtime.h>
#include <hip/hip_bf16.h>

// ---------------------------------------------------------------------------
// MoE: B=4, S=2048, D=1024, E=8, H=4096, TOP_K=2  (tokens T = 8192)
// Routed: gate -> per-expert lists -> bf16 WMMA GEMMs (f32 accum) -> combine.
// Weights/activations are pre-converted to bf16 (weights pre-transposed) so
// the GEMM hot loops stage LDS tiles with pure b128 copies (the gfx1250
// async-to-LDS path when available), double-buffered to overlap with WMMA.
// ---------------------------------------------------------------------------

#define T_TOKENS 8192
#define D_DIM    1024
#define E_EXP    8
#define H_DIM    4096
#define LD_K     72       // padded LDS row pitch (bf16 elements)

typedef __attribute__((ext_vector_type(16))) __bf16 v16bf;
typedef __attribute__((ext_vector_type(8)))  float  v8f;
typedef __attribute__((ext_vector_type(4)))  int    v4i;

#if __has_builtin(__builtin_amdgcn_global_load_async_to_lds_b128) && \
    __has_builtin(__builtin_amdgcn_s_wait_asynccnt)
#define USE_ASYNC 1
typedef __attribute__((address_space(1))) v4i gv4i;   // global
typedef __attribute__((address_space(3))) v4i lv4i;   // LDS
#else
#define USE_ASYNC 0
#endif

// 16-byte global->LDS copy (async-to-LDS when available)
static __device__ inline void cp16(__bf16* dst, const __bf16* src) {
#if USE_ASYNC
    __builtin_amdgcn_global_load_async_to_lds_b128(
        (gv4i*)(const_cast<__bf16*>(src)), (lv4i*)(dst), 0, 0);
#else
    *(uint4*)dst = *(const uint4*)src;
#endif
}

static __device__ inline void async_fence() {
#if USE_ASYNC
    __builtin_amdgcn_s_wait_asynccnt(0);
#endif
}

// 16 bf16 fragment from two 16B LDS chunks
static __device__ inline v16bf load_frag16(const __bf16* p0, const __bf16* p1) {
    union U { uint4 q[2]; v16bf v; } u;
    u.q[0] = *(const uint4*)p0;
    u.q[1] = *(const uint4*)p1;
    return u.v;
}

// ---------------- kernel 0: zero the per-expert counters --------------------
__global__ void moe_zero_counts(int* counts) {
    if (threadIdx.x < E_EXP) counts[threadIdx.x] = 0;
}

// ---------------- kernel 1: x fp32 -> bf16 ----------------------------------
__global__ void moe_cvt_x(const float* __restrict__ x, __bf16* __restrict__ xb) {
    int gid = blockIdx.x * blockDim.x + threadIdx.x;
    const float4 f = ((const float4*)x)[gid];
    union { __bf16 h[4]; uint2 u; } pk;
    pk.h[0] = (__bf16)f.x; pk.h[1] = (__bf16)f.y;
    pk.h[2] = (__bf16)f.z; pk.h[3] = (__bf16)f.w;
    ((uint2*)xb)[gid] = pk.u;
}

// ---------------- kernel 2: weight transpose + fp32 -> bf16 -----------------
// in[e]: R x C fp32 row-major   ->   out[e]: C x R bf16 row-major
__global__ void moe_transpose_cvt(const float* __restrict__ in,
                                  __bf16* __restrict__ out, int R, int C) {
    __shared__ float t[32][33];
    const size_t eo = (size_t)blockIdx.z * R * C;
    in  += eo;
    out += eo;
    const int c0 = blockIdx.x * 32, r0 = blockIdx.y * 32;
    const int tc = threadIdx.x & 31, tr = threadIdx.x >> 5;   // 32 x 8
#pragma unroll
    for (int i = 0; i < 4; ++i)
        t[tr + 8 * i][tc] = in[(size_t)(r0 + tr + 8 * i) * C + c0 + tc];
    __syncthreads();
#pragma unroll
    for (int i = 0; i < 4; ++i)
        out[(size_t)(c0 + tr + 8 * i) * R + r0 + tc] = (__bf16)t[tc][tr + 8 * i];
}

// ---------------- kernel 3: gating (one wave32 per token) -------------------
__global__ void moe_gate(const float* __restrict__ x,
                         const float* __restrict__ gate_w,
                         int* __restrict__ counts,
                         int* __restrict__ lists,     // [E][T] entries tok*2+k
                         float* __restrict__ wgt,     // [T][2]
                         int* __restrict__ sel_out) { // [T][2]
    int tok  = (blockIdx.x * blockDim.x + threadIdx.x) >> 5;
    int lane = threadIdx.x & 31;
    if (tok >= T_TOKENS) return;

    const float* xr = x + (size_t)tok * D_DIM;
    float acc[E_EXP];
#pragma unroll
    for (int e = 0; e < E_EXP; ++e) acc[e] = 0.f;

    for (int d = lane; d < D_DIM; d += 32) {
        float xv = xr[d];
        const float* g = gate_w + (size_t)d * E_EXP;
#pragma unroll
        for (int e = 0; e < E_EXP; ++e) acc[e] += xv * g[e];
    }
#pragma unroll
    for (int e = 0; e < E_EXP; ++e) {
#pragma unroll
        for (int off = 16; off > 0; off >>= 1)
            acc[e] += __shfl_xor(acc[e], off, 32);
    }

    if (lane == 0) {
        int   i0 = 0; float v0 = acc[0];
#pragma unroll
        for (int e = 1; e < E_EXP; ++e)
            if (acc[e] > v0) { v0 = acc[e]; i0 = e; }   // ties -> lowest index
        int   i1 = (i0 == 0) ? 1 : 0; float v1 = acc[i1];
#pragma unroll
        for (int e = 0; e < E_EXP; ++e)
            if (e != i0 && acc[e] > v1) { v1 = acc[e]; i1 = e; }

        float e1 = __expf(v1 - v0);
        float s  = 1.0f + e1;
        wgt[tok * 2 + 0] = 1.0f / s;
        wgt[tok * 2 + 1] = e1 / s;
        sel_out[tok * 2 + 0] = i0;
        sel_out[tok * 2 + 1] = i1;

        int p0 = atomicAdd(&counts[i0], 1);
        lists[i0 * T_TOKENS + p0] = tok * 2 + 0;
        int p1 = atomicAdd(&counts[i1], 1);
        lists[i1 * T_TOKENS + p1] = tok * 2 + 1;
    }
}

// ---------------- kernel 4: per-expert GEMM1 + bias + ReLU ------------------
// He[row, 0:H] = relu( gather(xb)[row] @ w1[e] + b1[e] ),  row < count[e]
// A: gathered xb rows (bf16), B: w1t[e] = w1[e]^T, [H][D] bf16.
__global__ __launch_bounds__(256)
void moe_gemm1(const __bf16* __restrict__ xb,
               const __bf16* __restrict__ w1t,
               const float*  __restrict__ b1e,
               const int*    __restrict__ count_p,
               const int*    __restrict__ list,
               __bf16*       __restrict__ He) {
    const int cnt  = *count_p;
    const int row0 = blockIdx.y * 128;
    if (row0 >= cnt) return;
    const int n0   = blockIdx.x * 128;

    __shared__ __bf16 lA[2][128 * LD_K];
    __shared__ __bf16 lB[2][128 * LD_K];

    const int tid  = threadIdx.x;
    const int lane = tid & 31;
    const int wv   = tid >> 5;
    const int wm   = wv >> 2;
    const int wn   = wv & 3;
    const int ml   = lane & 15;
    const int hh   = lane >> 4;

    v8f acc[4][2] = {};

    auto stage = [&](int k0, int buf) {
#pragma unroll
        for (int it = 0; it < 4; ++it) {           // A: 128x64 bf16, gathered
            int li  = (tid + it * 256) * 8;
            int r   = li >> 6, c = li & 63;
            int row = row0 + r;
            int rr  = row < cnt ? row : (cnt - 1);
            int tok = list[rr] >> 1;
            cp16(&lA[buf][r * LD_K + c], xb + (size_t)tok * D_DIM + k0 + c);
        }
#pragma unroll
        for (int it = 0; it < 4; ++it) {           // B: rows of w1^T, contiguous K
            int li = (tid + it * 256) * 8;
            int n  = li >> 6, kk = li & 63;
            cp16(&lB[buf][n * LD_K + kk], w1t + (size_t)(n0 + n) * D_DIM + k0 + kk);
        }
    };

    stage(0, 0);
    async_fence();
    __syncthreads();

    for (int k0 = 0; k0 < D_DIM; k0 += 64) {
        const int buf = (k0 >> 6) & 1;
        if (k0 + 64 < D_DIM) stage(k0 + 64, buf ^ 1);

#pragma unroll
        for (int ks = 0; ks < 2; ++ks) {
            v16bf af[4], bfr[2];
#pragma unroll
            for (int fm = 0; fm < 4; ++fm) {
                const __bf16* p = &lA[buf][(wm * 64 + fm * 16 + ml) * LD_K + ks * 32 + hh * 8];
                af[fm] = load_frag16(p, p + 16);
            }
#pragma unroll
            for (int fn = 0; fn < 2; ++fn) {
                const __bf16* p = &lB[buf][(wn * 32 + fn * 16 + ml) * LD_K + ks * 32 + hh * 8];
                bfr[fn] = load_frag16(p, p + 16);
            }
#pragma unroll
            for (int fm = 0; fm < 4; ++fm)
#pragma unroll
                for (int fn = 0; fn < 2; ++fn)
                    acc[fm][fn] = __builtin_amdgcn_wmma_f32_16x16x32_bf16(
                        false, af[fm], false, bfr[fn], (short)0, acc[fm][fn], false, false);
        }
        async_fence();
        __syncthreads();
    }

#pragma unroll
    for (int fm = 0; fm < 4; ++fm)
#pragma unroll
        for (int fn = 0; fn < 2; ++fn)
#pragma unroll
            for (int r = 0; r < 8; ++r) {
                int row = row0 + wm * 64 + fm * 16 + hh * 8 + r;
                int col = n0 + wn * 32 + fn * 16 + ml;
                if (row < cnt) {
                    float v = fmaxf(acc[fm][fn][r] + b1e[col], 0.0f);
                    He[(size_t)row * H_DIM + col] = (__bf16)v;
                }
            }
}

// ---------------- kernel 5: per-expert GEMM2 + bias + scatter ---------------
// ybuf[list[row]] = He[row] @ w2[e] + b2[e];  B: w2t[e] = w2[e]^T, [D][H] bf16.
__global__ __launch_bounds__(256)
void moe_gemm2(const __bf16* __restrict__ He,
               const __bf16* __restrict__ w2t,
               const float*  __restrict__ b2e,
               const int*    __restrict__ count_p,
               const int*    __restrict__ list,
               float*        __restrict__ ybuf) {
    const int cnt  = *count_p;
    const int row0 = blockIdx.y * 128;
    if (row0 >= cnt) return;
    const int n0   = blockIdx.x * 128;

    __shared__ __bf16 lA[2][128 * LD_K];
    __shared__ __bf16 lB[2][128 * LD_K];

    const int tid  = threadIdx.x;
    const int lane = tid & 31;
    const int wv   = tid >> 5;
    const int wm   = wv >> 2;
    const int wn   = wv & 3;
    const int ml   = lane & 15;
    const int hh   = lane >> 4;

    v8f acc[4][2] = {};

    auto stage = [&](int k0, int buf) {
#pragma unroll
        for (int it = 0; it < 4; ++it) {           // A: He rows (local, dense)
            int li  = (tid + it * 256) * 8;
            int r   = li >> 6, c = li & 63;
            int row = row0 + r;
            int rr  = row < cnt ? row : (cnt - 1);
            cp16(&lA[buf][r * LD_K + c], He + (size_t)rr * H_DIM + k0 + c);
        }
#pragma unroll
        for (int it = 0; it < 4; ++it) {           // B: rows of w2^T, contiguous K
            int li = (tid + it * 256) * 8;
            int n  = li >> 6, kk = li & 63;
            cp16(&lB[buf][n * LD_K + kk], w2t + (size_t)(n0 + n) * H_DIM + k0 + kk);
        }
    };

    stage(0, 0);
    async_fence();
    __syncthreads();

    for (int k0 = 0; k0 < H_DIM; k0 += 64) {
        const int buf = (k0 >> 6) & 1;
        if (k0 + 64 < H_DIM) stage(k0 + 64, buf ^ 1);

#pragma unroll
        for (int ks = 0; ks < 2; ++ks) {
            v16bf af[4], bfr[2];
#pragma unroll
            for (int fm = 0; fm < 4; ++fm) {
                const __bf16* p = &lA[buf][(wm * 64 + fm * 16 + ml) * LD_K + ks * 32 + hh * 8];
                af[fm] = load_frag16(p, p + 16);
            }
#pragma unroll
            for (int fn = 0; fn < 2; ++fn) {
                const __bf16* p = &lB[buf][(wn * 32 + fn * 16 + ml) * LD_K + ks * 32 + hh * 8];
                bfr[fn] = load_frag16(p, p + 16);
            }
#pragma unroll
            for (int fm = 0; fm < 4; ++fm)
#pragma unroll
                for (int fn = 0; fn < 2; ++fn)
                    acc[fm][fn] = __builtin_amdgcn_wmma_f32_16x16x32_bf16(
                        false, af[fm], false, bfr[fn], (short)0, acc[fm][fn], false, false);
        }
        async_fence();
        __syncthreads();
    }

#pragma unroll
    for (int fm = 0; fm < 4; ++fm)
#pragma unroll
        for (int fn = 0; fn < 2; ++fn)
#pragma unroll
            for (int r = 0; r < 8; ++r) {
                int row = row0 + wm * 64 + fm * 16 + hh * 8 + r;
                int col = n0 + wn * 32 + fn * 16 + ml;
                if (row < cnt) {
                    int entry = list[row];                  // tok*2 + slot
                    ybuf[(size_t)entry * D_DIM + col] = acc[fm][fn][r] + b2e[col];
                }
            }
}

// ---------------- kernel 6: deterministic weighted combine ------------------
__global__ void moe_combine(const float* __restrict__ ybuf,
                            const float* __restrict__ wgt,
                            float* __restrict__ out) {
    int gid = blockIdx.x * blockDim.x + threadIdx.x;
    if (gid >= T_TOKENS * D_DIM) return;
    int t = gid >> 10;
    int c = gid & (D_DIM - 1);
    out[gid] = wgt[t * 2 + 0] * ybuf[(size_t)(t * 2 + 0) * D_DIM + c]
             + wgt[t * 2 + 1] * ybuf[(size_t)(t * 2 + 1) * D_DIM + c];
}

// ---------------------------------------------------------------------------
extern "C" void kernel_launch(void* const* d_in, const int* in_sizes, int n_in,
                              void* d_out, int out_size, void* d_ws, size_t ws_size,
                              hipStream_t stream) {
    (void)in_sizes; (void)n_in; (void)out_size; (void)ws_size;

    const float* x      = (const float*)d_in[0];
    const float* gate_w = (const float*)d_in[1];
    const float* w1     = (const float*)d_in[2];
    const float* b1     = (const float*)d_in[3];
    const float* w2     = (const float*)d_in[4];
    const float* b2     = (const float*)d_in[5];

    float* out = (float*)d_out;
    int*   sel = (int*)d_out + (size_t)T_TOKENS * D_DIM;   // tuple output #2

    // workspace layout
    char*   ws     = (char*)d_ws;
    size_t  off    = 0;
    int*    counts = (int*)(ws + off);  off += 256;
    int*    lists  = (int*)(ws + off);  off += sizeof(int) * E_EXP * T_TOKENS;
    float*  wgt    = (float*)(ws + off); off += sizeof(float) * 2 * T_TOKENS;
    off = (off + 255) & ~(size_t)255;
    __bf16* xb     = (__bf16*)(ws + off); off += sizeof(__bf16) * (size_t)T_TOKENS * D_DIM;
    __bf16* w1t    = (__bf16*)(ws + off); off += sizeof(__bf16) * (size_t)E_EXP * D_DIM * H_DIM;
    __bf16* w2t    = (__bf16*)(ws + off); off += sizeof(__bf16) * (size_t)E_EXP * H_DIM * D_DIM;
    __bf16* He     = (__bf16*)(ws + off); off += sizeof(__bf16) * (size_t)T_TOKENS * H_DIM;
    float*  ybuf   = (float*)(ws + off);

    moe_zero_counts<<<1, 32, 0, stream>>>(counts);
    moe_cvt_x<<<(T_TOKENS * D_DIM / 4) / 256, 256, 0, stream>>>(x, xb);
    // w1[e]: [D,H] -> w1t[e]: [H,D]
    moe_transpose_cvt<<<dim3(H_DIM / 32, D_DIM / 32, E_EXP), 256, 0, stream>>>(w1, w1t, D_DIM, H_DIM);
    // w2[e]: [H,D] -> w2t[e]: [D,H]
    moe_transpose_cvt<<<dim3(D_DIM / 32, H_DIM / 32, E_EXP), 256, 0, stream>>>(w2, w2t, H_DIM, D_DIM);
    moe_gate<<<T_TOKENS / 4, 128, 0, stream>>>(x, gate_w, counts, lists, wgt, sel);

    for (int e = 0; e < E_EXP; ++e) {
        moe_gemm1<<<dim3(H_DIM / 128, T_TOKENS / 128), 256, 0, stream>>>(
            xb, w1t + (size_t)e * D_DIM * H_DIM, b1 + (size_t)e * H_DIM,
            counts + e, lists + (size_t)e * T_TOKENS, He);
        moe_gemm2<<<dim3(D_DIM / 128, T_TOKENS / 128), 256, 0, stream>>>(
            He, w2t + (size_t)e * H_DIM * D_DIM, b2 + (size_t)e * D_DIM,
            counts + e, lists + (size_t)e * T_TOKENS, ybuf);
    }

    moe_combine<<<(T_TOKENS * D_DIM) / 256, 256, 0, stream>>>(ybuf, wgt, out);
}
